// AttentionBlock_5669356832815
// MI455X (gfx1250) — matrix operand order repaired and checked by reading, not compiled, not run on hardware
//
#include <hip/hip_runtime.h>
#include <hip/hip_bf16.h>

typedef __attribute__((ext_vector_type(16))) __bf16 v16bf;
typedef __attribute__((ext_vector_type(8)))  __bf16 v8bf;
typedef __attribute__((ext_vector_type(8)))  float  v8f;
typedef __attribute__((ext_vector_type(4)))  unsigned int v4u;
typedef __attribute__((ext_vector_type(8)))  int v8i;
typedef __attribute__((ext_vector_type(4)))  int v4i;

union ABf { v16bf v; v8bf h[2]; };

#define Bv 8
#define Cv 512
#define Nv 4096
#define Gv 32
#define AT_I 64
#define AT_J 32

static __device__ __forceinline__ v8f wmma_bf16(v16bf a, v16bf b, v8f c) {
  // emits v_wmma_f32_16x16x32_bf16
  return __builtin_amdgcn_wmma_f32_16x16x32_bf16(false, a, false, b, (short)0, c, false, false);
}

// ---------------------------------------------------------------------------
// TDM 2-D tile load: Global -> LDS, bf16 elements (data_size=2B).
// Descriptor packing per CDNA5 ISA sec. 8.3/8.4 (D# groups 0 and 1; 2-D tile
// so groups 2/3 and the extra group are null). Issued by one wave; tracked
// with TENSORcnt. Toolchain uses the 6-arg builtin form.
// ---------------------------------------------------------------------------
static __device__ __forceinline__ void tdm_load_2d(unsigned int lds_off, const void* gptr,
                                                   unsigned int tile_d0, unsigned int tile_d1,
                                                   unsigned int tensor_d0, unsigned int tensor_d1,
                                                   unsigned long long stride0_elems) {
  unsigned long long ga = (unsigned long long)gptr;
  v4u g0;
  g0.x = 1u;                                                   // count=1 (valid user D#)
  g0.y = lds_off;                                              // lds_addr (bytes)
  g0.z = (unsigned int)ga;                                     // global_addr[31:0]
  g0.w = (unsigned int)((ga >> 32) & 0x1FFFFFFull) | (2u << 30); // addr[56:32] | type=2
  v8i g1;
  g1[0] = (int)(1u << 16);                                     // wg_mask=0, data_size=1 (2B)
  g1[1] = (int)((tensor_d0 & 0xFFFFu) << 16);                  // tensor_dim0[15:0] @ bits63:48
  g1[2] = (int)(((tensor_d0 >> 16) & 0xFFFFu) | ((tensor_d1 & 0xFFFFu) << 16));
  g1[3] = (int)(((tensor_d1 >> 16) & 0xFFFFu) | ((tile_d0 & 0xFFFFu) << 16));
  g1[4] = (int)(tile_d1 & 0xFFFFu);                            // tile_dim1; tile_dim2=0
  g1[5] = (int)(unsigned int)(stride0_elems & 0xFFFFFFFFull);  // dim0_stride[31:0]
  g1[6] = (int)((stride0_elems >> 32) & 0xFFFFull);            // dim0_stride[47:32]; dim1_stride=0
  g1[7] = 0;
  v4i gz4 = {0, 0, 0, 0};
  v8i gz8 = {0, 0, 0, 0, 0, 0, 0, 0};
  __builtin_amdgcn_tensor_load_to_lds(g0, g1, gz4, gz4, gz8, 0);
}

static __device__ __forceinline__ unsigned int lds_offset_of(const void* p) {
  // flat shared address: aperture in high 32 bits, LDS byte offset in low 32
  return (unsigned int)(unsigned long long)p;
}

// ---------------------------------------------------------------------------
// 1) Weight prep: fp32 -> bf16, fold 1/sqrt(C) into Wq/bq.
// ---------------------------------------------------------------------------
__global__ void prep_kernel(const float* __restrict__ Wq, const float* __restrict__ bq,
                            const float* __restrict__ Wk, const float* __restrict__ bk,
                            const float* __restrict__ Wv, const float* __restrict__ bv,
                            const float* __restrict__ Wp,
                            __bf16* __restrict__ wcat, __bf16* __restrict__ wpb,
                            float* __restrict__ bqkv) {
  const float s = 0.044194173824159216f; // 1/sqrt(512)
  int i = blockIdx.x * 256 + threadIdx.x;
  const int WSZ = Cv * Cv;
  if (i < 3 * WSZ) {
    int m = i >> 9;
    float v = (m < Cv) ? Wq[i] * s : (m < 2 * Cv ? Wk[i - WSZ] : Wv[i - 2 * WSZ]);
    wcat[i] = (__bf16)v;
  } else if (i < 4 * WSZ) {
    int j = i - 3 * WSZ;
    wpb[j] = (__bf16)Wp[j];
  } else if (i < 4 * WSZ + 3 * Cv) {
    int j = i - 4 * WSZ;
    bqkv[j] = (j < Cv) ? bq[j] * s : (j < 2 * Cv ? bk[j - Cv] : bv[j - 2 * Cv]);
  }
}

// ---------------------------------------------------------------------------
// 2) GroupNorm: one block per (batch, group); write bf16 token-major hnT[B,N,C]
// ---------------------------------------------------------------------------
__global__ __launch_bounds__(256) void groupnorm_kernel(const float* __restrict__ x,
                                                        const float* __restrict__ gamma,
                                                        const float* __restrict__ beta,
                                                        __bf16* __restrict__ hnT) {
  const int b = blockIdx.x >> 5, g = blockIdx.x & 31;
  const float* xg = x + ((size_t)b * Cv + g * 16) * Nv;  // 16 channels x 4096
  float s = 0.f, s2 = 0.f;
  for (int i = threadIdx.x; i < 16 * Nv; i += 256) { float v = xg[i]; s += v; s2 += v * v; }
  __shared__ float rs[256], rs2[256];
  rs[threadIdx.x] = s; rs2[threadIdx.x] = s2; __syncthreads();
  for (int off = 128; off > 0; off >>= 1) {
    if (threadIdx.x < off) { rs[threadIdx.x] += rs[threadIdx.x + off]; rs2[threadIdx.x] += rs2[threadIdx.x + off]; }
    __syncthreads();
  }
  const float inv_cnt = 1.f / (16.f * Nv);
  float mean = rs[0] * inv_cnt;
  float var  = rs2[0] * inv_cnt - mean * mean;
  float rstd = rsqrtf(var + 1e-5f);
  for (int i = threadIdx.x; i < 16 * Nv; i += 256) {
    int cl = i >> 12, n = i & (Nv - 1);
    int c = g * 16 + cl;
    float v = (xg[i] - mean) * rstd * gamma[c] + beta[c];
    hnT[((size_t)b * Nv + n) * Cv + c] = (__bf16)v;
  }
}

// ---------------------------------------------------------------------------
// 3) QKV GEMM, blocked: each wave computes a 64x32 output block
//    (4 m-tiles x 2 n-tiles) -> A-frags reused 2x, B-frags 4x from registers.
// ---------------------------------------------------------------------------
__global__ __launch_bounds__(32) void qkv_kernel(const __bf16* __restrict__ wcat,
                                                 const __bf16* __restrict__ hnT,
                                                 const float* __restrict__ bqkv,
                                                 __bf16* __restrict__ qT, __bf16* __restrict__ kT,
                                                 __bf16* __restrict__ vC) {
  const int b = blockIdx.z;
  const int mbase = blockIdx.y * 64;   // 0..1472
  const int nbase = blockIdx.x * 32;   // 0..4064
  const int lane = threadIdx.x, lo = lane & 15, hi = lane >> 4;
  const v8f vz = {0.f,0.f,0.f,0.f,0.f,0.f,0.f,0.f};
  v8f acc[4][2];
#pragma unroll
  for (int a = 0; a < 4; ++a)
#pragma unroll
    for (int nn = 0; nn < 2; ++nn) acc[a][nn] = vz;

  const __bf16* arow0 = wcat + (size_t)(mbase + lo) * Cv;
  const __bf16* brow0 = hnT + ((size_t)b * Nv + nbase + lo) * Cv;
#pragma unroll
  for (int ck = 0; ck < Cv / 32; ++ck) {
    ABf af[4], bf[2];
#pragma unroll
    for (int a = 0; a < 4; ++a) {
      const __bf16* p = arow0 + (size_t)a * 16 * Cv + ck * 32 + hi * 16;
      af[a].h[0] = *(const v8bf*)(p);
      af[a].h[1] = *(const v8bf*)(p + 8);
    }
#pragma unroll
    for (int nn = 0; nn < 2; ++nn)
      bf[nn].v = *(const v16bf*)(brow0 + (size_t)nn * 16 * Cv + ck * 32 + hi * 16);
#pragma unroll
    for (int a = 0; a < 4; ++a)
#pragma unroll
      for (int nn = 0; nn < 2; ++nn)
        acc[a][nn] = wmma_bf16(af[a].v, bf[nn].v, acc[a][nn]);
  }
#pragma unroll
  for (int a = 0; a < 4; ++a) {
#pragma unroll
    for (int nn = 0; nn < 2; ++nn) {
      int n = nbase + nn * 16 + lo;
#pragma unroll
      for (int v = 0; v < 8; ++v) {
        int m = mbase + a * 16 + v + hi * 8;
        float val = acc[a][nn][v] + bqkv[m];
        if (m < Cv)          qT[((size_t)b * Nv + n) * Cv + m]            = (__bf16)val;
        else if (m < 2 * Cv) kT[((size_t)b * Nv + n) * Cv + (m - Cv)]     = (__bf16)val;
        else                 vC[((size_t)b * Cv + (m - 2 * Cv)) * Nv + n] = (__bf16)val;
      }
    }
  }
}

// ---------------------------------------------------------------------------
// 4) Flash attention. 256 threads = 8 waves, i-tile 64, j-tile 32.
//    Q/K/V tiles staged via the Tensor Data Mover (TENSOR_LOAD_TO_LDS).
//    LDS: Qs 64KB + Ks 32KB + Vs 32KB + Ps 8KB = 136KB.
// ---------------------------------------------------------------------------
__global__ __launch_bounds__(256) void attn_kernel(const __bf16* __restrict__ qT,
                                                   const __bf16* __restrict__ kT,
                                                   const __bf16* __restrict__ vC,
                                                   __bf16* __restrict__ attT) {
  const int b = blockIdx.y;
  const int ibase = blockIdx.x * AT_I;
  const int tid = threadIdx.x;
  const int lane = tid & 31, lo = lane & 15, hi = lane >> 4;
  const int wid = tid >> 5, iw = wid & 3, cw = wid >> 2;

  __shared__ __bf16 Qs[AT_I][Cv];        // [i][c]
  __shared__ __bf16 Ks[AT_J][Cv];        // [j][c]
  __shared__ __bf16 Vs[Cv][AT_J];        // [c][j]
  __shared__ __bf16 Ps[8][16][AT_J];     // per-wave probabilities [i][j]

  const v8f vz = {0.f,0.f,0.f,0.f,0.f,0.f,0.f,0.f};
  v8f acc_o[16]; // 16 i x 256 c per wave
#pragma unroll
  for (int t = 0; t < 16; ++t) acc_o[t] = vz;
  float mrow[8], lrow[8];
#pragma unroll
  for (int v = 0; v < 8; ++v) { mrow[v] = -1e30f; lrow[v] = 0.f; }

  // Q tile via TDM: 64 rows x 512 cols, row stride C (contiguous rows)
  if (wid == 0) {
    tdm_load_2d(lds_offset_of(&Qs[0][0]), qT + ((size_t)b * Nv + ibase) * Cv,
                Cv, AT_I, Cv, Nv, Cv);
  }

  for (int js = 0; js < Nv / AT_J; ++js) {
    const int jbase = js * AT_J;
    __syncthreads();                     // prev iter compute done before overwrite
    if (wid == 0) {
      // K tile: 32 rows x 512 cols from token-major kT
      tdm_load_2d(lds_offset_of(&Ks[0][0]), kT + ((size_t)b * Nv + jbase) * Cv,
                  Cv, AT_J, Cv, Nv, Cv);
      // V tile: 512 rows x 32 cols from channel-major v (row stride N)
      tdm_load_2d(lds_offset_of(&Vs[0][0]), vC + (size_t)b * Cv * Nv + jbase,
                  AT_J, Cv, Nv, Cv, Nv);
      __builtin_amdgcn_s_wait_tensorcnt(0);  // DMA done (covers Q on first iter)
    }
    __syncthreads();                     // publish LDS tiles to all waves
    // Prefetch next tiles toward L2 (global_prefetch_b8)
    if (js + 1 < Nv / AT_J) {
      __builtin_prefetch(kT + ((size_t)b * Nv + jbase + AT_J + (tid >> 3)) * Cv + (tid & 7) * 64, 0, 0);
      __builtin_prefetch(vC + ((size_t)b * Cv + tid * 2) * Nv + jbase + AT_J, 0, 0);
    }

    // ---- scores S = Q_tile * K_tile^T : 16 i x 32 j per wave ----
    v8f acc_s[2] = {vz, vz};
#pragma unroll
    for (int ck = 0; ck < Cv / 32; ++ck) {
      ABf a;
      a.h[0] = *(const v8bf*)(&Qs[iw * 16 + lo][ck * 32 + hi * 16]);
      a.h[1] = *(const v8bf*)(&Qs[iw * 16 + lo][ck * 32 + hi * 16 + 8]);
#pragma unroll
      for (int jt = 0; jt < 2; ++jt) {
        ABf bm;
        bm.v = *(const v16bf*)(&Ks[jt * 16 + lo][ck * 32 + hi * 16]);
        acc_s[jt] = wmma_bf16(a.v, bm.v, acc_s[jt]);
      }
    }

    // ---- online softmax (row = v + 8*hi, col = lane%16 per D layout) ----
#pragma unroll
    for (int v = 0; v < 8; ++v) {
      float tm = fmaxf(acc_s[0][v], acc_s[1][v]);
#pragma unroll
      for (int off = 8; off >= 1; off >>= 1) tm = fmaxf(tm, __shfl_xor(tm, off, 32));
      float mn = fmaxf(mrow[v], tm);
      float f = __expf(mrow[v] - mn);
      mrow[v] = mn;
      float rs = 0.f;
#pragma unroll
      for (int t = 0; t < 2; ++t) {
        float p = __expf(acc_s[t][v] - mn);
        acc_s[t][v] = p;
        rs += p;
      }
#pragma unroll
      for (int off = 8; off >= 1; off >>= 1) rs += __shfl_xor(rs, off, 32);
      lrow[v] = lrow[v] * f + rs;
#pragma unroll
      for (int ct = 0; ct < 16; ++ct) acc_o[ct][v] *= f;
      int prow = v + hi * 8;
#pragma unroll
      for (int t = 0; t < 2; ++t)
        Ps[wid][prow][t * 16 + lo] = (__bf16)acc_s[t][v];
    }
    // own-wave LDS RAW across lanes: make the P stores visible
    asm volatile("s_wait_dscnt 0" ::: "memory");

    // ---- output accumulate: O += P * V^T (K-dim = 32 j) ----
    ABf pa;
    pa.h[0] = *(const v8bf*)(&Ps[wid][lo][hi * 16]);
    pa.h[1] = *(const v8bf*)(&Ps[wid][lo][hi * 16 + 8]);
#pragma unroll
    for (int ct = 0; ct < 16; ++ct) {
      int cc = cw * 256 + ct * 16 + lo;
      ABf bm;
      bm.v = *(const v16bf*)(&Vs[cc][hi * 16]);
      acc_o[ct] = wmma_bf16(pa.v, bm.v, acc_o[ct]);
    }
  }

  // ---- normalize and store att^T [B,N,C] bf16 ----
#pragma unroll
  for (int ct = 0; ct < 16; ++ct) {
    int c = cw * 256 + ct * 16 + lo;
#pragma unroll
    for (int v = 0; v < 8; ++v) {
      int i = ibase + iw * 16 + v + hi * 8;
      attT[((size_t)b * Nv + i) * Cv + c] = (__bf16)(acc_o[ct][v] / lrow[v]);
    }
  }
}

// ---------------------------------------------------------------------------
// 5) Output projection + residual, blocked 64x32 per wave: out = x + Wp*att + bp
// ---------------------------------------------------------------------------
__global__ __launch_bounds__(32) void proj_kernel(const __bf16* __restrict__ wpb,
                                                  const __bf16* __restrict__ attT,
                                                  const float* __restrict__ bp,
                                                  const float* __restrict__ x,
                                                  float* __restrict__ out) {
  const int b = blockIdx.z;
  const int obase = blockIdx.y * 64;
  const int nbase = blockIdx.x * 32;
  const int lane = threadIdx.x, lo = lane & 15, hi = lane >> 4;
  const v8f vz = {0.f,0.f,0.f,0.f,0.f,0.f,0.f,0.f};
  v8f acc[4][2];
#pragma unroll
  for (int a = 0; a < 4; ++a)
#pragma unroll
    for (int nn = 0; nn < 2; ++nn) acc[a][nn] = vz;

  const __bf16* arow0 = wpb + (size_t)(obase + lo) * Cv;
  const __bf16* brow0 = attT + ((size_t)b * Nv + nbase + lo) * Cv;
#pragma unroll
  for (int ck = 0; ck < Cv / 32; ++ck) {
    ABf af[4], bf[2];
#pragma unroll
    for (int a = 0; a < 4; ++a) {
      const __bf16* p = arow0 + (size_t)a * 16 * Cv + ck * 32 + hi * 16;
      af[a].h[0] = *(const v8bf*)(p);
      af[a].h[1] = *(const v8bf*)(p + 8);
    }
#pragma unroll
    for (int nn = 0; nn < 2; ++nn)
      bf[nn].v = *(const v16bf*)(brow0 + (size_t)nn * 16 * Cv + ck * 32 + hi * 16);
#pragma unroll
    for (int a = 0; a < 4; ++a)
#pragma unroll
      for (int nn = 0; nn < 2; ++nn)
        acc[a][nn] = wmma_bf16(af[a].v, bf[nn].v, acc[a][nn]);
  }
#pragma unroll
  for (int a = 0; a < 4; ++a) {
#pragma unroll
    for (int nn = 0; nn < 2; ++nn) {
      int n = nbase + nn * 16 + lo;
#pragma unroll
      for (int v = 0; v < 8; ++v) {
        int o = obase + a * 16 + v + hi * 8;
        size_t idx = ((size_t)b * Cv + o) * Nv + n;
        out[idx] = x[idx] + acc[a][nn][v] + bp[o];
      }
    }
  }
}

// ---------------------------------------------------------------------------
extern "C" void kernel_launch(void* const* d_in, const int* in_sizes, int n_in,
                              void* d_out, int out_size, void* d_ws, size_t ws_size,
                              hipStream_t stream) {
  const float* x     = (const float*)d_in[0];
  const float* gamma = (const float*)d_in[1];
  const float* beta  = (const float*)d_in[2];
  const float* Wq = (const float*)d_in[3]; const float* bq = (const float*)d_in[4];
  const float* Wk = (const float*)d_in[5]; const float* bk = (const float*)d_in[6];
  const float* Wv = (const float*)d_in[7]; const float* bv = (const float*)d_in[8];
  const float* Wp = (const float*)d_in[9]; const float* bp = (const float*)d_in[10];

  char* ws = (char*)d_ws;
  const size_t MB = 1024ull * 1024ull;
  const size_t ACT = (size_t)Bv * Nv * Cv * sizeof(__bf16);  // 32 MB each
  __bf16* hnT  = (__bf16*)(ws);
  __bf16* qT   = (__bf16*)(ws + 1 * ACT);
  __bf16* kT   = (__bf16*)(ws + 2 * ACT);
  __bf16* vC   = (__bf16*)(ws + 3 * ACT);
  __bf16* attT = (__bf16*)(ws + 4 * ACT);
  __bf16* wcat = (__bf16*)(ws + 5 * ACT);               // 1.5 MB
  __bf16* wpb  = (__bf16*)(ws + 5 * ACT + 2 * MB);      // 0.5 MB
  float*  bqkv = (float*) (ws + 5 * ACT + 3 * MB);      // 6 KB

  // 1) weights -> bf16 (score scale folded into Wq/bq)
  int prep_elems = 4 * Cv * Cv + 3 * Cv;
  prep_kernel<<<(prep_elems + 255) / 256, 256, 0, stream>>>(Wq, bq, Wk, bk, Wv, bv, Wp,
                                                            wcat, wpb, bqkv);
  // 2) GroupNorm -> hnT bf16 [B,N,C]
  groupnorm_kernel<<<Bv * Gv, 256, 0, stream>>>(x, gamma, beta, hnT);
  // 3) QKV projections (WMMA, 64x32 per wave)
  qkv_kernel<<<dim3(Nv / 32, 3 * Cv / 64, Bv), 32, 0, stream>>>(wcat, hnT, bqkv, qT, kT, vC);
  // 4) Flash attention (WMMA + TDM staging + online softmax)
  attn_kernel<<<dim3(Nv / AT_I, Bv), 256, 0, stream>>>(qT, kT, vC, attT);
  // 5) Output projection + residual (WMMA, fp32 out)
  proj_kernel<<<dim3(Nv / 32, Cv / 64, Bv), 32, 0, stream>>>(wpb, attT, bp, x, (float*)d_out);
}
